// Nequip_46145128628617
// MI455X (gfx1250) — matrix operand order repaired and checked by reading, not compiled, not run on hardware
//
#include <hip/hip_runtime.h>
#include <math.h>

#define NATOMS 30000
#define NEDGES 300000
#define NE2    600000
#define NTILES (NE2/16)
#define CCH    16
#define NLAY   3
#define NBAS   8
#define HID    64
#define WOUT   80

#define INV_RMAX   0.25f
#define DEG_NORM   0.223606797749979f   /* 1/sqrt(E2/N) = 1/sqrt(20) */
#define SQRT3C     1.7320508075688772f
#define INV_SQRT3  0.5773502691896258f
#define INV_SQRT2  0.7071067811865476f
#define RBF_COEF   0.7071067811865476f  /* sqrt(2/r_max) */

typedef __attribute__((ext_vector_type(16))) _Float16 v16h;
typedef __attribute__((ext_vector_type(8)))  float    v8f;

#define WMMA(a,b,c) __builtin_amdgcn_wmma_f32_16x16x32_f16(false,(a),false,(b),(short)0,(c),false,false)
#define DS_FENCE()  asm volatile("s_wait_dscnt 0" ::: "memory")

/* hardware v_rcp_f32: ~1ulp, no div_scale/div_fmas refinement chains */
__device__ __forceinline__ float frcp(float x) { return __builtin_amdgcn_rcpf(x); }

__device__ __forceinline__ v8f zero8() {
  v8f z;
#pragma unroll
  for (int i = 0; i < 8; ++i) z[i] = 0.f;
  return z;
}
__device__ __forceinline__ v16h zero16() {
  v16h z;
#pragma unroll
  for (int i = 0; i < 16; ++i) z[i] = (_Float16)0.f;
  return z;
}
__device__ __forceinline__ float siluf(float x) {
  float s = frcp(1.f + __expf(-x));
  return x * s;
}
__device__ __forceinline__ float dsiluf(float x) {
  float s = frcp(1.f + __expf(-x));
  return s * (1.f + x * (1.f - s));
}
__device__ __forceinline__ float denvf(float x, float r) {
  if (x < 1.f && r > 1e-6f) {
    float x2 = x * x, x4 = x2 * x2, x5 = x4 * x;
    return (-168.f * x5 + 336.f * x5 * x - 168.f * x5 * x2) * INV_RMAX;
  }
  return 0.f;
}
__device__ __forceinline__ void edge_sr(const int* __restrict__ ei, int e, int& snd, int& rcv) {
  rcv = ei[e];
  snd = (e < NEDGES) ? ei[e + NEDGES] : ei[e - NEDGES];
}

/* ---- WMMA fragment builders (contiguous k order, same for A and B) ---- */
/* A 16xK f16: lane L holds row M=L&15; half t -> K = kBase + 16*(L>>4) + t */
__device__ __forceinline__ v16h frag_A(const _Float16* m, int ld, int kBase, int lane, int Krows) {
  v16h a; int M = lane & 15; int hi = lane >> 4;
#pragma unroll
  for (int t = 0; t < 16; ++t) {
    int k = kBase + 16 * hi + t;
    a[t] = (k < Krows) ? m[M * ld + k] : (_Float16)0.f;
  }
  return a;
}
/* A with on-the-fly SiLU (stage holds pre-activations) */
__device__ __forceinline__ v16h frag_A_silu(const _Float16* m, int ld, int kBase, int lane) {
  v16h a; int M = lane & 15; int hi = lane >> 4;
#pragma unroll
  for (int t = 0; t < 16; ++t) {
    int k = kBase + 16 * hi + t;
    float x = (float)m[M * ld + k];
    a[t] = (_Float16)siluf(x);
  }
  return a;
}
/* B 32x16 f16: lane L holds col N=nBase+(L&15); half t -> K = kBase + 16*(L>>4) + t */
__device__ __forceinline__ v16h frag_B(const _Float16* m, int ld, int kBase, int nBase, int lane, int Krows) {
  v16h b; int n = nBase + (lane & 15); int k0 = kBase + ((lane >> 4) << 4);
#pragma unroll
  for (int t = 0; t < 16; ++t) {
    int k = k0 + t;
    b[t] = (k < Krows) ? m[k * ld + n] : (_Float16)0.f;
  }
  return b;
}
/* Transposed-weight B: B[k][n] = m[n*ld + k] (m row-major [Ncols][ld]) */
__device__ __forceinline__ v16h frag_BT(const _Float16* m, int ld, int kBase, int nBase, int lane,
                                        int Krows, int Ncols) {
  v16h b; int n = nBase + (lane & 15); int k0 = kBase + ((lane >> 4) << 4);
  bool nok = (n < Ncols);
#pragma unroll
  for (int t = 0; t < 16; ++t) {
    int k = k0 + t;
    b[t] = (nok && k < Krows) ? m[n * ld + k] : (_Float16)0.f;
  }
  return b;
}

/* =================== geometry =================== */
__global__ void k_geom(const float* __restrict__ pos, const float* __restrict__ cell,
                       const int* __restrict__ eidx, float* __restrict__ dvec,
                       float* __restrict__ rarr, float* __restrict__ envarr) {
  int e = blockIdx.x * blockDim.x + threadIdx.x;
  if (e >= NE2) return;
  int snd, rcv; edge_sr(eidx, e, snd, rcv);
  float d[3], r2 = 1e-12f;
#pragma unroll
  for (int i = 0; i < 3; ++i) {
    float di = pos[snd * 3 + i] - pos[rcv * 3 + i];
    float ci = cell[i];
    di -= ci * rintf(di * frcp(ci));
    d[i] = di; r2 += di * di;
  }
  float r = sqrtf(r2);
  float x = r * INV_RMAX;
  float env = 0.f;
  if (x < 1.f && r > 1e-6f) {
    float x2 = x * x, x4 = x2 * x2, x6 = x4 * x2;
    env = 1.f - 28.f * x6 + 48.f * x6 * x - 21.f * x6 * x2;
  }
  dvec[e * 3 + 0] = d[0]; dvec[e * 3 + 1] = d[1]; dvec[e * 3 + 2] = d[2];
  rarr[e] = r; envarr[e] = env;
}

/* =================== node kernels =================== */
__global__ void k_embed(const float* __restrict__ Wemb, const int* __restrict__ at,
                        float* __restrict__ s) {
  int t = blockIdx.x * blockDim.x + threadIdx.x;
  if (t >= NATOMS * CCH) return;
  s[t] = Wemb[at[t >> 4] * CCH + (t & 15)];
}

__global__ void k_node_lin(const float* __restrict__ sIn, const float* __restrict__ vIn,
                           float* __restrict__ sOut, float* __restrict__ vOut,
                           const float* __restrict__ Ws, const float* __restrict__ Wv) {
  int t = blockIdx.x * blockDim.x + threadIdx.x;
  if (t >= NATOMS * CCH) return;
  int n = t >> 4, d = t & 15;
  float as = 0.f, a0 = 0.f, a1 = 0.f, a2 = 0.f;
#pragma unroll
  for (int c = 0; c < CCH; ++c) {
    float ws = Ws[c * CCH + d], wv = Wv[c * CCH + d];
    as += sIn[n * CCH + c] * ws;
    const float* vp = &vIn[(n * CCH + c) * 3];
    a0 += vp[0] * wv; a1 += vp[1] * wv; a2 += vp[2] * wv;
  }
  sOut[t] = as; vOut[t * 3 + 0] = a0; vOut[t * 3 + 1] = a1; vOut[t * 3 + 2] = a2;
}

__global__ void k_node_update(float* __restrict__ s, float* __restrict__ v,
                              const float* __restrict__ as_, const float* __restrict__ av_,
                              const float* __restrict__ Ws2, const float* __restrict__ Wv2) {
  int t = blockIdx.x * blockDim.x + threadIdx.x;
  if (t >= NATOMS * CCH) return;
  int n = t >> 4, d = t & 15;
  float as = 0.f, a0 = 0.f, a1 = 0.f, a2 = 0.f;
#pragma unroll
  for (int c = 0; c < CCH; ++c) {
    float ws = Ws2[c * CCH + d], wv = Wv2[c * CCH + d];
    as += as_[n * CCH + c] * ws;
    const float* vp = &av_[(n * CCH + c) * 3];
    a0 += vp[0] * wv; a1 += vp[1] * wv; a2 += vp[2] * wv;
  }
  s[t] += as; v[t * 3 + 0] += a0; v[t * 3 + 1] += a1; v[t * 3 + 2] += a2;
}

/* ga = g @ W^T  (out index c, contract over d) */
__global__ void k_node_gout(const float* __restrict__ gs, const float* __restrict__ gv,
                            float* __restrict__ gas, float* __restrict__ gav,
                            const float* __restrict__ Ws2, const float* __restrict__ Wv2) {
  int t = blockIdx.x * blockDim.x + threadIdx.x;
  if (t >= NATOMS * CCH) return;
  int n = t >> 4, c = t & 15;
  float as = 0.f, a0 = 0.f, a1 = 0.f, a2 = 0.f;
#pragma unroll
  for (int d = 0; d < CCH; ++d) {
    float ws = Ws2[c * CCH + d], wv = Wv2[c * CCH + d];
    as += gs[n * CCH + d] * ws;
    const float* vp = &gv[(n * CCH + d) * 3];
    a0 += vp[0] * wv; a1 += vp[1] * wv; a2 += vp[2] * wv;
  }
  gas[t] = as; gav[t * 3 + 0] = a0; gav[t * 3 + 1] = a1; gav[t * 3 + 2] = a2;
}

__global__ void k_node_gin(float* __restrict__ gs, float* __restrict__ gv,
                           const float* __restrict__ gs1, const float* __restrict__ gv1,
                           const float* __restrict__ Ws1, const float* __restrict__ Wv1) {
  int t = blockIdx.x * blockDim.x + threadIdx.x;
  if (t >= NATOMS * CCH) return;
  int n = t >> 4, c = t & 15;
  float as = 0.f, a0 = 0.f, a1 = 0.f, a2 = 0.f;
#pragma unroll
  for (int d = 0; d < CCH; ++d) {
    float ws = Ws1[c * CCH + d], wv = Wv1[c * CCH + d];
    as += gs1[n * CCH + d] * ws;
    const float* vp = &gv1[(n * CCH + d) * 3];
    a0 += vp[0] * wv; a1 += vp[1] * wv; a2 += vp[2] * wv;
  }
  gs[t] += as; gv[t * 3 + 0] += a0; gv[t * 3 + 1] += a1; gv[t * 3 + 2] += a2;
}

__global__ void k_energy(const float* __restrict__ s, const float* __restrict__ Whs,
                         const float* __restrict__ WE, const float* __restrict__ bE,
                         float* __restrict__ out) {
  int n = blockIdx.x * blockDim.x + threadIdx.x;
  if (n >= NATOMS) return;
  float e = bE[0];
#pragma unroll
  for (int d = 0; d < CCH; ++d) {
    float so = 0.f;
#pragma unroll
    for (int c = 0; c < CCH; ++c) so += s[n * CCH + c] * Whs[c * CCH + d];
    e += so * WE[d];
  }
  atomicAdd(out, e);
}

__global__ void k_init_grad(float* __restrict__ gs, const float* __restrict__ Whs,
                            const float* __restrict__ WE) {
  int t = blockIdx.x * blockDim.x + threadIdx.x;
  if (t >= NATOMS * CCH) return;
  int c = t & 15;
  float g = 0.f;
#pragma unroll
  for (int d = 0; d < CCH; ++d) g += Whs[c * CCH + d] * WE[d];
  gs[t] = g;
}

/* =================== fused forward edge kernel (WMMA) =================== */
__global__ __launch_bounds__(256) void k_edge_fwd(
    const float* __restrict__ rW0, const float* __restrict__ rb0,
    const float* __restrict__ rW1, const float* __restrict__ rb1,
    const float* __restrict__ rW2, const float* __restrict__ bw,
    const int* __restrict__ eidx, const float* __restrict__ dvec,
    const float* __restrict__ rarr, const float* __restrict__ envarr,
    const float* __restrict__ s1, const float* __restrict__ v1,
    float* __restrict__ as_, float* __restrict__ av_) {
  __shared__ _Float16 W0s[NBAS * HID];
  __shared__ _Float16 W1s[HID * HID];
  __shared__ _Float16 W2s[HID * WOUT];
  __shared__ float B0s[HID], B1s[HID];
  __shared__ _Float16 hst[8][16 * HID];

  for (int i = threadIdx.x; i < NBAS * HID; i += 256) W0s[i] = (_Float16)rW0[i];
  for (int i = threadIdx.x; i < HID * HID; i += 256) W1s[i] = (_Float16)rW1[i];
  for (int i = threadIdx.x; i < HID * WOUT; i += 256) W2s[i] = (_Float16)rW2[i];
  for (int i = threadIdx.x; i < HID; i += 256) { B0s[i] = rb0[i]; B1s[i] = rb1[i]; }
  __syncthreads();

  const int lane = threadIdx.x & 31;
  const int wid = threadIdx.x >> 5;
  const int hi = lane >> 4;
  const int lc = lane & 15;
  _Float16* hs = hst[wid];
  const int waves = gridDim.x * 8;

  for (int tile = blockIdx.x * 8 + wid; tile < NTILES; tile += waves) {
    const int base = tile * 16;
    /* rbf A-fragment: only K<8 nonzero -> lanes>=16 all zero */
    v16h a0 = zero16();
    if (lane < 16) {
      int e = base + lane;
      float r = rarr[e], env = envarr[e];
      float x = r * INV_RMAX, invr = frcp(r);
      float cf = RBF_COEF * env * invr;
#pragma unroll
      for (int n = 0; n < NBAS; ++n) a0[n] = (_Float16)(cf * __sinf(x * bw[n]));
    }
    /* h1 = silu(rbf @ W0 + b0) */
    v8f acc[4];
#pragma unroll
    for (int nt = 0; nt < 4; ++nt)
      acc[nt] = WMMA(a0, frag_B(W0s, HID, 0, nt * 16, lane, NBAS), zero8());
#pragma unroll
    for (int nt = 0; nt < 4; ++nt)
#pragma unroll
      for (int j = 0; j < 8; ++j) {
        int M = j + 8 * hi, col = nt * 16 + lc;
        hs[M * HID + col] = (_Float16)siluf(acc[nt][j] + B0s[col]);
      }
    DS_FENCE();
    /* h2 = silu(h1 @ W1 + b1) */
    v16h ak0 = frag_A(hs, HID, 0, lane, HID), ak1 = frag_A(hs, HID, 32, lane, HID);
#pragma unroll
    for (int nt = 0; nt < 4; ++nt) {
      v8f c = WMMA(ak0, frag_B(W1s, HID, 0, nt * 16, lane, HID), zero8());
      acc[nt] = WMMA(ak1, frag_B(W1s, HID, 32, nt * 16, lane, HID), c);
    }
#pragma unroll
    for (int nt = 0; nt < 4; ++nt)
#pragma unroll
      for (int j = 0; j < 8; ++j) {
        int M = j + 8 * hi, col = nt * 16 + lc;
        hs[M * HID + col] = (_Float16)siluf(acc[nt][j] + B1s[col]);
      }
    DS_FENCE();
    /* raw = h2 @ W2, 5 n-tiles (= 5 TP paths x 16 channels) */
    v16h bk0 = frag_A(hs, HID, 0, lane, HID), bk1 = frag_A(hs, HID, 32, lane, HID);
    v8f raw[5];
#pragma unroll
    for (int p = 0; p < 5; ++p) {
      v8f c = WMMA(bk0, frag_B(W2s, WOUT, 0, p * 16, lane, HID), zero8());
      raw[p] = WMMA(bk1, frag_B(W2s, WOUT, 32, p * 16, lane, HID), c);
    }
    /* messages: each lane = channel lc x 8 edge rows */
#pragma unroll
    for (int j = 0; j < 8; ++j) {
      int M = j + 8 * hi, e = base + M;
      int snd, rcv; edge_sr(eidx, e, snd, rcv);
      float env = envarr[e], r = rarr[e], invr = frcp(r);
      float Y0 = SQRT3C * dvec[e * 3 + 0] * invr;
      float Y1v = SQRT3C * dvec[e * 3 + 1] * invr;
      float Y2 = SQRT3C * dvec[e * 3 + 2] * invr;
      float sj = s1[snd * CCH + lc];
      const float* vp = &v1[(snd * CCH + lc) * 3];
      float vj0 = vp[0], vj1 = vp[1], vj2 = vp[2];
      float vdY = vj0 * Y0 + vj1 * Y1v + vj2 * Y2;
      float w0 = raw[0][j] * env, w1 = raw[1][j] * env, w2 = raw[2][j] * env;
      float w3 = raw[3][j] * env, w4 = raw[4][j] * env;
      float ms = DEG_NORM * (w0 * sj + w3 * vdY * INV_SQRT3);
      float cx0 = vj1 * Y2 - vj2 * Y1v, cx1 = vj2 * Y0 - vj0 * Y2, cx2 = vj0 * Y1v - vj1 * Y0;
      float mv0 = DEG_NORM * (w1 * sj * Y0 + w2 * vj0 + w4 * cx0 * INV_SQRT2);
      float mv1 = DEG_NORM * (w1 * sj * Y1v + w2 * vj1 + w4 * cx1 * INV_SQRT2);
      float mv2 = DEG_NORM * (w1 * sj * Y2 + w2 * vj2 + w4 * cx2 * INV_SQRT2);
      atomicAdd(&as_[rcv * CCH + lc], ms);
      float* ap = &av_[(rcv * CCH + lc) * 3];
      atomicAdd(&ap[0], mv0); atomicAdd(&ap[1], mv1); atomicAdd(&ap[2], mv2);
    }
  }
}

/* =================== fused backward edge kernel (WMMA fwd-recompute + bwd) =================== */
__global__ __launch_bounds__(128) void k_edge_bwd(
    const float* __restrict__ rW0, const float* __restrict__ rb0,
    const float* __restrict__ rW1, const float* __restrict__ rb1,
    const float* __restrict__ rW2, const float* __restrict__ bw,
    const int* __restrict__ eidx, const float* __restrict__ dvec,
    const float* __restrict__ rarr, const float* __restrict__ envarr,
    const float* __restrict__ s1, const float* __restrict__ v1,
    const float* __restrict__ gas, const float* __restrict__ gav,
    float* __restrict__ gs1, float* __restrict__ gv1, float* __restrict__ force) {
  __shared__ _Float16 W0s[NBAS * HID];
  __shared__ _Float16 W1s[HID * HID];
  __shared__ _Float16 W2s[HID * WOUT];
  __shared__ float B0s[HID], B1s[HID];
  __shared__ _Float16 h1p[4][16 * HID];   /* pre-activations, layer 1 */
  __shared__ _Float16 h2p[4][16 * HID];   /* pre-activations, layer 2 */
  __shared__ _Float16 stA[4][16 * WOUT];  /* g_raw / gh1pre staging */
  __shared__ _Float16 stB[4][16 * HID];   /* gh2pre staging */
  __shared__ float gYt[4][48], grt[4][16], gevt[4][16];

  for (int i = threadIdx.x; i < NBAS * HID; i += 128) W0s[i] = (_Float16)rW0[i];
  for (int i = threadIdx.x; i < HID * HID; i += 128) W1s[i] = (_Float16)rW1[i];
  for (int i = threadIdx.x; i < HID * WOUT; i += 128) W2s[i] = (_Float16)rW2[i];
  for (int i = threadIdx.x; i < HID; i += 128) { B0s[i] = rb0[i]; B1s[i] = rb1[i]; }
  __syncthreads();

  const int lane = threadIdx.x & 31;
  const int wid = threadIdx.x >> 5;
  const int hi = lane >> 4;
  const int lc = lane & 15;
  _Float16* p1 = h1p[wid];
  _Float16* p2 = h2p[wid];
  _Float16* sA = stA[wid];
  _Float16* sB = stB[wid];
  const int waves = gridDim.x * 4;

  for (int tile = blockIdx.x * 4 + wid; tile < NTILES; tile += waves) {
    const int base = tile * 16;
    if (lane < 16) {
      grt[wid][lane] = 0.f; gevt[wid][lane] = 0.f;
      gYt[wid][lane * 3 + 0] = 0.f; gYt[wid][lane * 3 + 1] = 0.f; gYt[wid][lane * 3 + 2] = 0.f;
    }
    DS_FENCE();
    /* ---- forward recompute ---- */
    v16h a0 = zero16();
    if (lane < 16) {
      int e = base + lane;
      float r = rarr[e], env = envarr[e];
      float x = r * INV_RMAX, invr = frcp(r);
      float cf = RBF_COEF * env * invr;
#pragma unroll
      for (int n = 0; n < NBAS; ++n) a0[n] = (_Float16)(cf * __sinf(x * bw[n]));
    }
    v8f acc[4];
#pragma unroll
    for (int nt = 0; nt < 4; ++nt)
      acc[nt] = WMMA(a0, frag_B(W0s, HID, 0, nt * 16, lane, NBAS), zero8());
#pragma unroll
    for (int nt = 0; nt < 4; ++nt)
#pragma unroll
      for (int j = 0; j < 8; ++j) {
        int M = j + 8 * hi, col = nt * 16 + lc;
        p1[M * HID + col] = (_Float16)(acc[nt][j] + B0s[col]);
      }
    DS_FENCE();
    v16h ak0 = frag_A_silu(p1, HID, 0, lane), ak1 = frag_A_silu(p1, HID, 32, lane);
#pragma unroll
    for (int nt = 0; nt < 4; ++nt) {
      v8f c = WMMA(ak0, frag_B(W1s, HID, 0, nt * 16, lane, HID), zero8());
      acc[nt] = WMMA(ak1, frag_B(W1s, HID, 32, nt * 16, lane, HID), c);
    }
#pragma unroll
    for (int nt = 0; nt < 4; ++nt)
#pragma unroll
      for (int j = 0; j < 8; ++j) {
        int M = j + 8 * hi, col = nt * 16 + lc;
        p2[M * HID + col] = (_Float16)(acc[nt][j] + B1s[col]);
      }
    DS_FENCE();
    v16h bk0 = frag_A_silu(p2, HID, 0, lane), bk1 = frag_A_silu(p2, HID, 32, lane);
    v8f raw[5];
#pragma unroll
    for (int p = 0; p < 5; ++p) {
      v8f c = WMMA(bk0, frag_B(W2s, WOUT, 0, p * 16, lane, HID), zero8());
      raw[p] = WMMA(bk1, frag_B(W2s, WOUT, 32, p * 16, lane, HID), c);
    }
    /* ---- message backward ---- */
#pragma unroll
    for (int j = 0; j < 8; ++j) {
      int M = j + 8 * hi, e = base + M;
      int snd, rcv; edge_sr(eidx, e, snd, rcv);
      float env = envarr[e], r = rarr[e], invr = frcp(r);
      float Y0 = SQRT3C * dvec[e * 3 + 0] * invr;
      float Y1v = SQRT3C * dvec[e * 3 + 1] * invr;
      float Y2 = SQRT3C * dvec[e * 3 + 2] * invr;
      float sj = s1[snd * CCH + lc];
      const float* vp = &v1[(snd * CCH + lc) * 3];
      float vj0 = vp[0], vj1 = vp[1], vj2 = vp[2];
      float vdY = vj0 * Y0 + vj1 * Y1v + vj2 * Y2;
      float w0 = raw[0][j] * env, w1 = raw[1][j] * env, w2 = raw[2][j] * env;
      float w3 = raw[3][j] * env, w4 = raw[4][j] * env;
      float gms = DEG_NORM * gas[rcv * CCH + lc];
      const float* gp = &gav[(rcv * CCH + lc) * 3];
      float g0 = DEG_NORM * gp[0], g1 = DEG_NORM * gp[1], g2v = DEG_NORM * gp[2];
      float gmvY = g0 * Y0 + g1 * Y1v + g2v * Y2;
      float cx0 = vj1 * Y2 - vj2 * Y1v, cx1 = vj2 * Y0 - vj0 * Y2, cx2 = vj0 * Y1v - vj1 * Y0;
      float gw0 = gms * sj;
      float gw1 = sj * gmvY;
      float gw2 = g0 * vj0 + g1 * vj1 + g2v * vj2;
      float gw3 = gms * vdY * INV_SQRT3;
      float gw4 = INV_SQRT2 * (g0 * cx0 + g1 * cx1 + g2v * cx2);
      float gvdY = gms * w3 * INV_SQRT3;
      float gsj = gms * w0 + w1 * gmvY;
      /* gvj = w2*gmv + gvdY*Y + (w4/sqrt2)*(Y x gmv) */
      float yg0 = Y1v * g2v - Y2 * g1, yg1 = Y2 * g0 - Y0 * g2v, yg2 = Y0 * g1 - Y1v * g0;
      float gvj0 = w2 * g0 + gvdY * Y0 + w4 * INV_SQRT2 * yg0;
      float gvj1 = w2 * g1 + gvdY * Y1v + w4 * INV_SQRT2 * yg1;
      float gvj2 = w2 * g2v + gvdY * Y2 + w4 * INV_SQRT2 * yg2;
      /* gY partial (channel-summed in LDS): w1*sj*gmv + gvdY*vj + (w4/sqrt2)*(gmv x vj) */
      float gv0 = g1 * vj2 - g2v * vj1, gv1c = g2v * vj0 - g0 * vj2, gv2 = g0 * vj1 - g1 * vj0;
      atomicAdd(&gYt[wid][M * 3 + 0], w1 * sj * g0 + gvdY * vj0 + w4 * INV_SQRT2 * gv0);
      atomicAdd(&gYt[wid][M * 3 + 1], w1 * sj * g1 + gvdY * vj1 + w4 * INV_SQRT2 * gv1c);
      atomicAdd(&gYt[wid][M * 3 + 2], w1 * sj * g2v + gvdY * vj2 + w4 * INV_SQRT2 * gv2);
      atomicAdd(&gevt[wid][M], gw0 * raw[0][j] + gw1 * raw[1][j] + gw2 * raw[2][j] +
                                   gw3 * raw[3][j] + gw4 * raw[4][j]);
      sA[M * WOUT + 0 * 16 + lc] = (_Float16)(gw0 * env);
      sA[M * WOUT + 1 * 16 + lc] = (_Float16)(gw1 * env);
      sA[M * WOUT + 2 * 16 + lc] = (_Float16)(gw2 * env);
      sA[M * WOUT + 3 * 16 + lc] = (_Float16)(gw3 * env);
      sA[M * WOUT + 4 * 16 + lc] = (_Float16)(gw4 * env);
      atomicAdd(&gs1[snd * CCH + lc], gsj);
      float* gq = &gv1[(snd * CCH + lc) * 3];
      atomicAdd(&gq[0], gvj0); atomicAdd(&gq[1], gvj1); atomicAdd(&gq[2], gvj2);
    }
    DS_FENCE();
    /* ---- gh2pre = (g_raw @ W2^T) * silu'(h2pre) ---- */
    v16h ga0 = frag_A(sA, WOUT, 0, lane, WOUT);
    v16h ga1 = frag_A(sA, WOUT, 32, lane, WOUT);
    v16h ga2 = frag_A(sA, WOUT, 64, lane, WOUT);
    v8f gcc[4];
#pragma unroll
    for (int nt = 0; nt < 4; ++nt) {
      v8f c = WMMA(ga0, frag_BT(W2s, WOUT, 0, nt * 16, lane, WOUT, HID), zero8());
      c = WMMA(ga1, frag_BT(W2s, WOUT, 32, nt * 16, lane, WOUT, HID), c);
      gcc[nt] = WMMA(ga2, frag_BT(W2s, WOUT, 64, nt * 16, lane, WOUT, HID), c);
    }
#pragma unroll
    for (int nt = 0; nt < 4; ++nt)
#pragma unroll
      for (int j = 0; j < 8; ++j) {
        int M = j + 8 * hi, col = nt * 16 + lc;
        sB[M * HID + col] = (_Float16)(gcc[nt][j] * dsiluf((float)p2[M * HID + col]));
      }
    DS_FENCE();
    /* ---- gh1pre = (gh2pre @ W1^T) * silu'(h1pre) ---- */
    v16h gb0 = frag_A(sB, HID, 0, lane, HID), gb1 = frag_A(sB, HID, 32, lane, HID);
#pragma unroll
    for (int nt = 0; nt < 4; ++nt) {
      v8f c = WMMA(gb0, frag_BT(W1s, HID, 0, nt * 16, lane, HID, HID), zero8());
      gcc[nt] = WMMA(gb1, frag_BT(W1s, HID, 32, nt * 16, lane, HID, HID), c);
    }
#pragma unroll
    for (int nt = 0; nt < 4; ++nt)
#pragma unroll
      for (int j = 0; j < 8; ++j) {
        int M = j + 8 * hi, col = nt * 16 + lc;
        sA[M * HID + col] = (_Float16)(gcc[nt][j] * dsiluf((float)p1[M * HID + col]));
      }
    DS_FENCE();
    /* ---- grbf = gh1pre @ W0^T (N=8 padded to 16) ---- */
    v16h gc0 = frag_A(sA, HID, 0, lane, HID), gc1 = frag_A(sA, HID, 32, lane, HID);
    v8f grb = WMMA(gc0, frag_BT(W0s, HID, 0, 0, lane, HID, NBAS), zero8());
    grb = WMMA(gc1, frag_BT(W0s, HID, 32, 0, lane, HID, NBAS), grb);
    if (lc < NBAS) {
      float wn = bw[lc];
#pragma unroll
      for (int j = 0; j < 8; ++j) {
        int M = j + 8 * hi, e = base + M;
        float r = rarr[e], env = envarr[e], invr = frcp(r);
        float x = r * INV_RMAX;
        float u = x * wn;
        float sn = __sinf(u), cs = __cosf(u);
        float dnv = denvf(x, r);
        float drbf = RBF_COEF * (env * (wn * INV_RMAX * cs * invr - sn * invr * invr) + dnv * sn * invr);
        atomicAdd(&grt[wid][M], grb[j] * drbf);
      }
    }
    DS_FENCE();
    /* ---- finalize per-edge geometric grad -> forces ---- */
    if (lane < 16) {
      int row = lane, e = base + row;
      float r = rarr[e], invr = frcp(r), x = r * INV_RMAX;
      float dnv = denvf(x, r);
      float gr = grt[wid][row] + gevt[wid][row] * dnv;
      float gY0 = gYt[wid][row * 3 + 0], gY1v = gYt[wid][row * 3 + 1], gY2 = gYt[wid][row * 3 + 2];
      float d0 = dvec[e * 3 + 0], d1 = dvec[e * 3 + 1], d2 = dvec[e * 3 + 2];
      float dgY = d0 * gY0 + d1 * gY1v + d2 * gY2;
      float k1 = SQRT3C * invr, k2 = SQRT3C * dgY * invr * invr * invr;
      float gd0 = gr * d0 * invr + k1 * gY0 - k2 * d0;
      float gd1 = gr * d1 * invr + k1 * gY1v - k2 * d1;
      float gd2 = gr * d2 * invr + k1 * gY2 - k2 * d2;
      int snd, rcv; edge_sr(eidx, e, snd, rcv);
      atomicAdd(&force[snd * 3 + 0], -gd0); atomicAdd(&force[snd * 3 + 1], -gd1);
      atomicAdd(&force[snd * 3 + 2], -gd2);
      atomicAdd(&force[rcv * 3 + 0], gd0); atomicAdd(&force[rcv * 3 + 1], gd1);
      atomicAdd(&force[rcv * 3 + 2], gd2);
    }
    DS_FENCE();
  }
}

/* =================== host =================== */
extern "C" void kernel_launch(void* const* d_in, const int* in_sizes, int n_in,
                              void* d_out, int out_size, void* d_ws, size_t ws_size,
                              hipStream_t stream) {
  (void)in_sizes; (void)n_in; (void)ws_size;
  const float* pos = (const float*)d_in[0];
  const float* cell = (const float*)d_in[1];
  const float* bw = (const float*)d_in[2];
  const float* Wemb = (const float*)d_in[3];
  const float* rW0 = (const float*)d_in[4];
  const float* rb0 = (const float*)d_in[5];
  const float* rW1 = (const float*)d_in[6];
  const float* rb1 = (const float*)d_in[7];
  const float* rW2 = (const float*)d_in[8];
  const float* Ws1 = (const float*)d_in[9];
  const float* Wv1 = (const float*)d_in[10];
  const float* Ws2 = (const float*)d_in[11];
  const float* Wv2 = (const float*)d_in[12];
  const float* Whs = (const float*)d_in[13];
  const float* WE = (const float*)d_in[14];
  const float* bE = (const float*)d_in[15];
  const int* eidx = (const int*)d_in[16];
  const int* atyp = (const int*)d_in[17];
  float* out = (float*)d_out;

  const size_t NC = (size_t)NATOMS * CCH;
  const size_t NC3 = NC * 3;
  float* W = (float*)d_ws;
  float* dvec = W;   W += (size_t)NE2 * 3;
  float* rarr = W;   W += NE2;
  float* envarr = W; W += NE2;
  float* s_cur = W;  W += NC;
  float* v_cur = W;  W += NC3;
  float* s_hist = W; W += NC * NLAY;
  float* v_hist = W; W += NC3 * NLAY;
  float* s1 = W;     W += NC;
  float* v1 = W;     W += NC3;
  float* asb = W;    W += NC;
  float* avb = W;    W += NC3;
  float* gs = W;     W += NC;
  float* gv = W;     W += NC3;
  float* gs1 = W;    W += NC;
  float* gv1 = W;    W += NC3;

  const int TPB = 256;
  const int gNC = (int)((NC + TPB - 1) / TPB);
  const int gN = (NATOMS + TPB - 1) / TPB;
  const int gE = (NE2 + TPB - 1) / TPB;

  hipMemsetAsync(d_out, 0, (size_t)out_size * sizeof(float), stream);
  hipMemsetAsync(v_cur, 0, NC3 * sizeof(float), stream);

  k_geom<<<gE, TPB, 0, stream>>>(pos, cell, eidx, dvec, rarr, envarr);
  k_embed<<<gNC, TPB, 0, stream>>>(Wemb, atyp, s_cur);

  /* ---------- forward layers ---------- */
  for (int l = 0; l < NLAY; ++l) {
    hipMemcpyAsync(s_hist + (size_t)l * NC, s_cur, NC * sizeof(float),
                   hipMemcpyDeviceToDevice, stream);
    hipMemcpyAsync(v_hist + (size_t)l * NC3, v_cur, NC3 * sizeof(float),
                   hipMemcpyDeviceToDevice, stream);
    k_node_lin<<<gNC, TPB, 0, stream>>>(s_cur, v_cur, s1, v1,
                                        Ws1 + l * CCH * CCH, Wv1 + l * CCH * CCH);
    hipMemsetAsync(asb, 0, NC * sizeof(float), stream);
    hipMemsetAsync(avb, 0, NC3 * sizeof(float), stream);
    k_edge_fwd<<<1024, 256, 0, stream>>>(rW0 + l * NBAS * HID, rb0 + l * HID,
                                         rW1 + l * HID * HID, rb1 + l * HID,
                                         rW2 + l * HID * WOUT, bw, eidx, dvec, rarr, envarr,
                                         s1, v1, asb, avb);
    k_node_update<<<gNC, TPB, 0, stream>>>(s_cur, v_cur, asb, avb,
                                           Ws2 + l * CCH * CCH, Wv2 + l * CCH * CCH);
  }

  /* ---------- energy + grad seed ---------- */
  k_energy<<<gN, TPB, 0, stream>>>(s_cur, Whs, WE, bE, out);
  k_init_grad<<<gNC, TPB, 0, stream>>>(gs, Whs, WE);
  hipMemsetAsync(gv, 0, NC3 * sizeof(float), stream);

  /* ---------- backward layers ---------- */
  for (int l = NLAY - 1; l >= 0; --l) {
    k_node_lin<<<gNC, TPB, 0, stream>>>(s_hist + (size_t)l * NC, v_hist + (size_t)l * NC3,
                                        s1, v1, Ws1 + l * CCH * CCH, Wv1 + l * CCH * CCH);
    k_node_gout<<<gNC, TPB, 0, stream>>>(gs, gv, asb, avb,
                                         Ws2 + l * CCH * CCH, Wv2 + l * CCH * CCH);
    hipMemsetAsync(gs1, 0, NC * sizeof(float), stream);
    hipMemsetAsync(gv1, 0, NC3 * sizeof(float), stream);
    k_edge_bwd<<<1024, 128, 0, stream>>>(rW0 + l * NBAS * HID, rb0 + l * HID,
                                         rW1 + l * HID * HID, rb1 + l * HID,
                                         rW2 + l * HID * WOUT, bw, eidx, dvec, rarr, envarr,
                                         s1, v1, asb, avb, gs1, gv1, out + 1);
    k_node_gin<<<gNC, TPB, 0, stream>>>(gs, gv, gs1, gv1,
                                        Ws1 + l * CCH * CCH, Wv1 + l * CCH * CCH);
  }
}
